// SHSA_12214886990429
// MI455X (gfx1250) — matrix operand order repaired and checked by reading, not compiled, not run on hardware
//
#include <hip/hip_runtime.h>
#include <hip/hip_bf16.h>
#include <stdint.h>

// ---------------------------------------------------------------------------
// SHSA block for MI455X (gfx1250, wave32, WMMA 16x16x32 bf16 / f32 acc)
// B=8, DIM=512, PDIM=256, QK=16, HW=4096
// ---------------------------------------------------------------------------

typedef __bf16 bf16_t;
typedef __attribute__((ext_vector_type(16))) __bf16 v16bf;
typedef __attribute__((ext_vector_type(8)))  float  v8f;

#define NB    8
#define DIM   512
#define PDIM  256
#define QKD   16
#define HW    4096
#define OQKV  288   // 2*QK + PDIM

// fast SiLU: hardware v_rcp_f32 instead of IEEE divide sequence
__device__ __forceinline__ float fast_silu(float v) {
    return v * __builtin_amdgcn_rcpf(1.f + __expf(-v));
}

// ---------------------------------------------------------------------------
// K1: GroupNorm stats: one block per batch, reduce sum/sumsq over PDIM*HW
// ---------------------------------------------------------------------------
__global__ void k_gn_stats(const float* __restrict__ x, float* __restrict__ stats) {
    const int b   = blockIdx.x;
    const int tid = threadIdx.x;
    const float* xb = x + (size_t)b * DIM * HW;
    float s = 0.f, ss = 0.f;
    for (int i = tid; i < PDIM * HW; i += 256) {
        float v = xb[i];
        s += v; ss += v * v;
    }
    __shared__ float rs_[256], rss_[256];
    rs_[tid] = s; rss_[tid] = ss;
    __syncthreads();
    for (int off = 128; off > 0; off >>= 1) {
        if (tid < off) { rs_[tid] += rs_[tid + off]; rss_[tid] += rss_[tid + off]; }
        __syncthreads();
    }
    if (tid == 0) {
        const float inv_n = 1.0f / (float)(PDIM * HW);
        float mu  = rs_[0] * inv_n;
        float var = rss_[0] * inv_n - mu * mu;
        stats[b * 2 + 0] = mu;
        stats[b * 2 + 1] = rsqrtf(var + 1e-5f);
    }
}

// ---------------------------------------------------------------------------
// K2a: apply GN affine, emit bf16 x1n [B][PDIM][HW]
// ---------------------------------------------------------------------------
__global__ void k_gn_apply(const float* __restrict__ x,
                           const float* __restrict__ gn_w,
                           const float* __restrict__ gn_b,
                           const float* __restrict__ stats,
                           bf16_t* __restrict__ x1n) {
    unsigned i = blockIdx.x * 256u + threadIdx.x;   // over NB*PDIM*HW = 2^23
    int b = i >> 20;
    int c = (i >> 12) & 255;
    int n = i & 4095;
    float mu = stats[b * 2 + 0];
    float rs = stats[b * 2 + 1];
    float v  = x[(size_t)b * DIM * HW + (size_t)c * HW + n];
    x1n[i] = (bf16_t)(((v - mu) * rs) * gn_w[c] + gn_b[c]);
}

// ---------------------------------------------------------------------------
// K2b: silu(x2) -> rows [PDIM..DIM) of y buffer [B][DIM][HW] bf16
// ---------------------------------------------------------------------------
__global__ void k_silu_x2(const float* __restrict__ x, bf16_t* __restrict__ y) {
    unsigned i = blockIdx.x * 256u + threadIdx.x;   // over NB*PDIM*HW
    int b = i >> 20;
    int c = (i >> 12) & 255;
    int n = i & 4095;
    float v = x[(size_t)b * DIM * HW + (size_t)(PDIM + c) * HW + n];
    y[(size_t)b * DIM * HW + (size_t)(PDIM + c) * HW + n] = (bf16_t)fast_silu(v);
}

// ---------------------------------------------------------------------------
// K2c: convert weights to bf16
// ---------------------------------------------------------------------------
__global__ void k_cvt_w(const float* __restrict__ wqkv, const float* __restrict__ wproj,
                        bf16_t* __restrict__ wq, bf16_t* __restrict__ wp) {
    unsigned i = blockIdx.x * 256u + threadIdx.x;
    if (i < OQKV * PDIM) wq[i] = (bf16_t)wqkv[i];
    if (i < DIM * DIM)   wp[i] = (bf16_t)wproj[i];
}

// ---------------------------------------------------------------------------
// Tiled bf16 WMMA GEMM:  Out[b][o][n] = sum_c A[o][c] * Bg[b][c][n] + bias[o]
// Block: 256 threads (8 waves); covers (OT*16) o-rows x 128 n-cols.
// Each wave owns 16 n-cols; one (expensive) B-fragment gather feeds OT WMMAs.
// ---------------------------------------------------------------------------
template <int OT, bool STORE_BF16>
__global__ __launch_bounds__(256) void k_gemm(
        const bf16_t* __restrict__ A,     // [O][K]
        const bf16_t* __restrict__ Bg,    // [NB][K][HW]
        const float*  __restrict__ bias,  // [O]
        void* __restrict__ Out,           // bf16 or f32 [NB][O][HW]
        int O, int K) {
    const int n0   = blockIdx.x * 128;
    const int o0   = blockIdx.y * (OT * 16);
    const int b    = blockIdx.z;
    const int tid  = threadIdx.x;
    const int wave = tid >> 5;
    const int lane = tid & 31;

    __shared__ bf16_t sA[OT * 16 * 32];
    __shared__ bf16_t sB[32 * 128];

    const bf16_t* bsrc = Bg + (size_t)b * K * HW;

    v8f acc[OT] = {};
    for (int k0 = 0; k0 < K; k0 += 32) {
        // --- stage A tile (OT*512 elems as dwords, OT/thread) ---
        #pragma unroll
        for (int j = 0; j < OT; ++j) {
            int e   = (tid + j * 256) * 2;
            int row = e >> 5, col = e & 31;
            *(uint32_t*)&sA[e] =
                *(const uint32_t*)&A[(size_t)(o0 + row) * K + k0 + col];
        }
        // --- stage B tile (4096 elems as 2048 dwords, 8/thread) ---
        #pragma unroll
        for (int j = 0; j < 8; ++j) {
            int e  = (tid + j * 256) * 2;
            int kk = e >> 7, nn = e & 127;
            *(uint32_t*)&sB[e] =
                *(const uint32_t*)&bsrc[(size_t)(k0 + kk) * HW + n0 + nn];
        }
        // --- prefetch next k-step's B rows into cache (global_prefetch_b8) ---
        if (k0 + 32 < K && tid < 64) {
            __builtin_prefetch(
                &bsrc[(size_t)(k0 + 32 + (tid >> 1)) * HW + n0 + (tid & 1) * 64], 0, 1);
        }
        __syncthreads();

        // --- B fragment (32x16): N across lanes, lane-half selects K+0/K+16 ---
        v16bf bfr;
        {
            const int N    = (lane & 15) + wave * 16;
            const int koff = (lane < 16) ? 0 : 16;
            #pragma unroll
            for (int r = 0; r < 8; ++r) {
                int kb = koff + 2 * r;
                bfr[2 * r]     = sB[kb * 128 + N];
                bfr[2 * r + 1] = sB[(kb + 1) * 128 + N];
            }
        }
        // --- A fragments + WMMA, OT output tiles share the B fragment ---
        const int M     = lane & 15;
    const int koffA = (lane < 16) ? 0 : 16;
        #pragma unroll
        for (int t = 0; t < OT; ++t) {
            v16bf af;
            #pragma unroll
            for (int r = 0; r < 8; ++r) {
                int kb = koffA + 2 * r;
                af[2 * r]     = sA[(t * 16 + M) * 32 + kb];
                af[2 * r + 1] = sA[(t * 16 + M) * 32 + kb + 1];
            }
            acc[t] = __builtin_amdgcn_wmma_f32_16x16x32_bf16(
                         false, af, false, bfr, (short)0, acc[t], false, false);
        }
        __syncthreads();
    }

    // --- store D tiles: M = r + lane-half*8, N = wave's 16-col strip ---
    const int N = n0 + wave * 16 + (lane & 15);
    #pragma unroll
    for (int t = 0; t < OT; ++t) {
        #pragma unroll
        for (int r = 0; r < 8; ++r) {
            int M = o0 + t * 16 + r + ((lane < 16) ? 0 : 8);
            float v = acc[t][r] + bias[M];
            if (STORE_BF16)
                ((bf16_t*)Out)[((size_t)b * O + M) * HW + N] = (bf16_t)v;
            else
                ((float*)Out)[((size_t)b * O + M) * HW + N] = v;
        }
    }
}

// ---------------------------------------------------------------------------
// K4: attention. One block per (batch, 32-col n-tile).
// Full softmax row set S[32][4096] = 256 KB LDS (CDNA5: 320 KB/WGP), so the
// 4096x4096 attention matrix never touches HBM, and the 32-wide n-tile halves
// the V re-read L2 traffic vs a 16-wide tile.
// x1o = V * P^T via bf16 WMMA: each wave does 2x2 tiles (4 WMMA) per m-step
// from 2 A-gathers (global b128 pairs) + 2 B-gathers (LDS b128 pairs).
// SiLU fused into the y-store.
// ---------------------------------------------------------------------------
__global__ __launch_bounds__(256) void k_attn(
        const bf16_t* __restrict__ qkv,   // [NB][OQKV][HW]
        bf16_t* __restrict__ y) {         // [NB][DIM][HW], rows 0..PDIM-1 written
    const int b    = blockIdx.y;
    const int n0   = blockIdx.x * 32;
    const int tid  = threadIdx.x;
    const int wave = tid >> 5;
    const int lane = tid & 31;

    __shared__ bf16_t sP[32 * HW];        // 256 KB: logits, then softmaxed P
    __shared__ float  sQ[16 * 32];        // q tile (c-major)
    __shared__ float  red[32 * 8];        // per-row reduction scratch

    const bf16_t* qb = qkv + (size_t)b * OQKV * HW;            // q rows 0..15
    const bf16_t* kb = qb + (size_t)QKD * HW;                  // k rows 16..31
    const bf16_t* vb = qb + (size_t)(2 * QKD) * HW;            // v rows 32..287

    // q tile: sQ[c*32 + j] = q[c][n0+j], 512 elems
    {
        int i = tid;
        sQ[i] = (float)qb[(size_t)(i >> 5) * HW + n0 + (i & 31)];
        i += 256;
        sQ[i] = (float)qb[(size_t)(i >> 5) * HW + n0 + (i & 31)];
    }
    __syncthreads();

    // --- logits: S[j][m] = scale * sum_c q[c][n0+j] k[c][m]  (K-dim only 16) ---
    const float scale = 0.25f;            // QK^-0.5
    for (int m = tid; m < HW; m += 256) {
        float s[32];
        #pragma unroll
        for (int j = 0; j < 32; ++j) s[j] = 0.f;
        #pragma unroll
        for (int c = 0; c < 16; ++c) {
            float kv = (float)kb[(size_t)c * HW + m];
            #pragma unroll
            for (int j = 0; j < 32; ++j) s[j] += sQ[c * 32 + j] * kv;
        }
        #pragma unroll
        for (int j = 0; j < 32; ++j) sP[j * HW + m] = (bf16_t)(s[j] * scale);
    }
    __syncthreads();

    // --- softmax over m, 8 threads per row (32 rows) ---
    {
        const int r = tid >> 3, p = tid & 7;
        float mx = -1e30f;
        for (int m = p; m < HW; m += 8) mx = fmaxf(mx, (float)sP[r * HW + m]);
        red[r * 8 + p] = mx;
        __syncthreads();
        if (p == 0) {
            float m2 = red[r * 8];
            for (int i = 1; i < 8; ++i) m2 = fmaxf(m2, red[r * 8 + i]);
            red[r * 8] = m2;
        }
        __syncthreads();
        const float rowmax = red[r * 8];
        __syncthreads();
        float sum = 0.f;
        for (int m = p; m < HW; m += 8) {
            float e = __expf((float)sP[r * HW + m] - rowmax);
            sum += e;
            sP[r * HW + m] = (bf16_t)e;
        }
        red[r * 8 + p] = sum;
        __syncthreads();
        if (p == 0) {
            float s2 = 0.f;
            for (int i = 0; i < 8; ++i) s2 += red[r * 8 + i];
            red[r * 8] = s2;
        }
        __syncthreads();
        const float inv = __builtin_amdgcn_rcpf(red[r * 8]);
        for (int m = p; m < HW; m += 8)
            sP[r * HW + m] = (bf16_t)((float)sP[r * HW + m] * inv);
    }
    __syncthreads();

    // --- x1o = V(256 x 4096) * P^T(4096 x 32); wave owns c-rows [32w, 32w+32) ---
    v8f acc[2][2] = {};                   // [c-tile][n-tile]
    const int Ma    = lane & 15;
    const int koffA = (lane < 16) ? 0 : 16;
    const int koffB = (lane < 16) ? 0 : 16;
    const int Nc    = lane & 15;
    const bf16_t* vrow0 = vb + (size_t)(wave * 32 + Ma) * HW;
    const bf16_t* vrow1 = vrow0 + (size_t)16 * HW;

    for (int m0 = 0; m0 < HW; m0 += 32) {
        v16bf pf0, pf1, a0, a1;
        uint32_t* pf0u = (uint32_t*)&pf0;
        uint32_t* pf1u = (uint32_t*)&pf1;
        uint32_t* a0u  = (uint32_t*)&a0;
        uint32_t* a1u  = (uint32_t*)&a1;
        #pragma unroll
        for (int r = 0; r < 8; ++r) {
            int kkB = m0 + koffB + 2 * r;             // P^T: K pairs contiguous
            pf0u[r] = *(const uint32_t*)&sP[(size_t)Nc * HW + kkB];
            pf1u[r] = *(const uint32_t*)&sP[(size_t)(16 + Nc) * HW + kkB];
            int kkA = m0 + koffA + 2 * r;             // A: K pairs contiguous
            a0u[r]  = *(const uint32_t*)&vrow0[kkA];
            a1u[r]  = *(const uint32_t*)&vrow1[kkA];
        }
        acc[0][0] = __builtin_amdgcn_wmma_f32_16x16x32_bf16(
                        false, a0, false, pf0, (short)0, acc[0][0], false, false);
        acc[0][1] = __builtin_amdgcn_wmma_f32_16x16x32_bf16(
                        false, a0, false, pf1, (short)0, acc[0][1], false, false);
        acc[1][0] = __builtin_amdgcn_wmma_f32_16x16x32_bf16(
                        false, a1, false, pf0, (short)0, acc[1][0], false, false);
        acc[1][1] = __builtin_amdgcn_wmma_f32_16x16x32_bf16(
                        false, a1, false, pf1, (short)0, acc[1][1], false, false);
    }

    // --- fused SiLU + store to y rows 0..255 ---
    bf16_t* yb = y + (size_t)b * DIM * HW;
    #pragma unroll
    for (int ct = 0; ct < 2; ++ct) {
        #pragma unroll
        for (int nt = 0; nt < 2; ++nt) {
            const int N = n0 + nt * 16 + (lane & 15);
            #pragma unroll
            for (int r = 0; r < 8; ++r) {
                int c = wave * 32 + ct * 16 + r + ((lane < 16) ? 0 : 8);
                yb[(size_t)c * HW + N] = (bf16_t)fast_silu(acc[ct][nt][r]);
            }
        }
    }
}

// ---------------------------------------------------------------------------
// Launcher
// ---------------------------------------------------------------------------
extern "C" void kernel_launch(void* const* d_in, const int* in_sizes, int n_in,
                              void* d_out, int out_size, void* d_ws, size_t ws_size,
                              hipStream_t stream) {
    const float* x      = (const float*)d_in[0];
    const float* gn_w   = (const float*)d_in[1];
    const float* gn_b   = (const float*)d_in[2];
    const float* w_qkv  = (const float*)d_in[3];
    const float* b_qkv  = (const float*)d_in[4];
    const float* w_proj = (const float*)d_in[5];
    const float* b_proj = (const float*)d_in[6];
    float* out = (float*)d_out;

    char* ws = (char*)d_ws;
    size_t off = 0;
    float*  stats = (float*)(ws + off);  off += 256;
    bf16_t* x1n   = (bf16_t*)(ws + off); off += (size_t)NB * PDIM * HW * 2;   // 16 MB
    bf16_t* wq    = (bf16_t*)(ws + off); off += (size_t)OQKV * PDIM * 2 + 256;
    bf16_t* wp    = (bf16_t*)(ws + off); off += (size_t)DIM * DIM * 2;
    bf16_t* qkvb  = (bf16_t*)(ws + off); off += (size_t)NB * OQKV * HW * 2;   // ~19 MB
    bf16_t* ybuf  = (bf16_t*)(ws + off); off += (size_t)NB * DIM * HW * 2;    // ~34 MB

    // 1) GN stats
    k_gn_stats<<<NB, 256, 0, stream>>>(x, stats);
    // 2) prep: GN apply, silu(x2), weight conversion
    const int NEL = NB * PDIM * HW;                   // 8,388,608
    k_gn_apply<<<NEL / 256, 256, 0, stream>>>(x, gn_w, gn_b, stats, x1n);
    k_silu_x2<<<NEL / 256, 256, 0, stream>>>(x, ybuf);
    k_cvt_w<<<(DIM * DIM + 255) / 256, 256, 0, stream>>>(w_qkv, w_proj, wq, wp);
    // 3) QKV GEMM (288x256 @ 256x4096 per batch), bias fused, bf16 out
    k_gemm<2, true><<<dim3(HW / 128, OQKV / 32, NB), 256, 0, stream>>>(
        wq, x1n, b_qkv, (void*)qkvb, OQKV, PDIM);
    // 4) attention + fused SiLU into y rows 0..255
    k_attn<<<dim3(HW / 32, NB), 256, 0, stream>>>(qkvb, ybuf);
    // 5) projection GEMM (512x512 @ 512x4096 per batch), bias fused, f32 out
    k_gemm<4, false><<<dim3(HW / 128, DIM / 64, NB), 256, 0, stream>>>(
        wp, ybuf, b_proj, (void*)out, DIM, DIM);
}
